// TwoTowerFirstStagePolicy_15049565405474
// MI455X (gfx1250) — matrix operand order repaired and checked by reading, not compiled, not run on hardware
//
#include <hip/hip_runtime.h>
#include <math.h>
#include <stdint.h>

#define BATCH   1024
#define DIMC    256
#define EMB     128
#define NITEMS  250000
#define TOPK    100
#define CHUNK   2048
#define NCHUNK  ((NITEMS + CHUNK - 1) / CHUNK)   // 123
#define ROWS_PER_BLOCK 32
#define MTILES  (ROWS_PER_BLOCK / 16)            // 2
#define ROWBLOCKS (BATCH / ROWS_PER_BLOCK)       // 32

#if __has_builtin(__builtin_amdgcn_global_load_async_to_lds_b128) && \
    __has_builtin(__builtin_amdgcn_s_wait_asynccnt)
#define HAS_ASYNC 1
#else
#define HAS_ASYNC 0
#endif

typedef __attribute__((ext_vector_type(16))) __bf16  bf16x16;
typedef __attribute__((ext_vector_type(8)))  float   floatx8;
typedef __attribute__((ext_vector_type(4)))  int     v4i;

// ---------- helpers ----------

__device__ __forceinline__ unsigned short f2bf_bits(float f) {
  unsigned u = __float_as_uint(f);
  u += 0x7FFFu + ((u >> 16) & 1u);          // round-to-nearest-even
  return (unsigned short)(u >> 16);
}

__device__ __forceinline__ float bf2f(unsigned short s) {
  return __uint_as_float((unsigned)s << 16);
}

// monotone order-preserving key for f32
__device__ __forceinline__ unsigned fkey(float f) {
  unsigned u = __float_as_uint(f);
  return (u & 0x80000000u) ? ~u : (u | 0x80000000u);
}

// monotone 16-bit key for bf16 bits
__device__ __forceinline__ unsigned key16(unsigned short s) {
  return (s & 0x8000u) ? (unsigned)((~s) & 0xFFFFu) : (unsigned)(s | 0x8000u);
}

__device__ __forceinline__ floatx8 wmma_bf16(bf16x16 a, bf16x16 b, floatx8 c) {
  return __builtin_amdgcn_wmma_f32_16x16x32_bf16(
      false, a, false, b, (short)0, c, false, false);
}

// A fragment from row-major bf16 (global or LDS): two b128 loads per lane.
__device__ __forceinline__ bf16x16 load_afrag_h(const unsigned short* src, int ld, int kb, int lane) {
  const int l  = lane & 15;
  const int hi = (lane >> 4) & 1;
  const unsigned short* p = src + (size_t)l * ld + kb + hi * 8;
  union { uint4 q[2]; bf16x16 v; } u;
  u.q[0] = *(const uint4*)p;          // K kb+hi*8 .. +7
  u.q[1] = *(const uint4*)(p + 16);   // K kb+16+hi*8 .. +7
  return u.v;
}

// B fragment from an N-major bf16 table (tab[n*ld + k]): lane n holds row nb+n,
// 16 k values (kb+hi*8..+7, kb+16+hi*8..+7) -> two b128 loads.
__device__ __forceinline__ bf16x16 load_bfrag_T(const unsigned short* tab, int ld, int nb,
                                                int kb, int lane, int maxrow) {
  const int l  = lane & 15;
  const int hi = (lane >> 4) & 1;
  int row = nb + l;
  if (row >= maxrow) row = maxrow - 1;
  const unsigned short* p = tab + (size_t)row * ld + kb + hi * 8;
  union { uint4 q[2]; bf16x16 v; } u;
  u.q[0] = *(const uint4*)p;
  u.q[1] = *(const uint4*)(p + 16);
  return u.v;
}

#if HAS_ASYNC
typedef __attribute__((address_space(1))) v4i* as1_v4ip;
typedef __attribute__((address_space(3))) v4i* as3_v4ip;

// Stage 16 contiguous item rows (4 KB of bf16) into LDS: 8 x async b128 per lane.
__device__ __forceinline__ void stage_tile_async(const unsigned short* embh, int nb,
                                                 unsigned short* dstLds, int lane) {
  const char* g = (const char*)(embh + (size_t)nb * EMB) + lane * 16;
  char*       d = (char*)dstLds + lane * 16;
#pragma unroll
  for (int j = 0; j < 8; ++j) {
    __builtin_amdgcn_global_load_async_to_lds_b128(
        (as1_v4ip)(uintptr_t)(g + j * 512),
        (as3_v4ip)(unsigned)(uintptr_t)(d + j * 512),
        0, 0);
  }
}

// B fragment from a staged 16x128 bf16 tile in LDS.
__device__ __forceinline__ bf16x16 load_bfrag_lds(const unsigned short* st, int kb, int lane) {
  const int l  = lane & 15;
  const int hi = (lane >> 4) & 1;
  const unsigned short* p = st + l * EMB + kb + hi * 8;
  union { uint4 q[2]; bf16x16 v; } u;
  u.q[0] = *(const uint4*)p;
  u.q[1] = *(const uint4*)(p + 16);
  return u.v;
}
#endif

// ---------- kernel 0a: f32 -> bf16 conversion of the item table ----------

__device__ __forceinline__ unsigned pack_bf16(float lo, float hi) {
  return (unsigned)f2bf_bits(lo) | ((unsigned)f2bf_bits(hi) << 16);
}

__global__ __launch_bounds__(256) void cvt_emb_kernel(const float* __restrict__ emb,
                                                      unsigned short* __restrict__ embh) {
  const size_t i = ((size_t)blockIdx.x * 256 + threadIdx.x) * 8;
  if (i + 8 > (size_t)NITEMS * EMB) return;
  float4 f0 = *(const float4*)(emb + i);
  float4 f1 = *(const float4*)(emb + i + 4);
  uint4 o;
  o.x = pack_bf16(f0.x, f0.y);
  o.y = pack_bf16(f0.z, f0.w);
  o.z = pack_bf16(f1.x, f1.y);
  o.w = pack_bf16(f1.z, f1.w);
  *(uint4*)(embh + i) = o;
}

// ---------- kernel 0b: convert x -> bf16; transpose W1, W2 -> N-major bf16 ----------

__global__ __launch_bounds__(256) void cvt_params_kernel(
    const float* __restrict__ x, const float* __restrict__ W1, const float* __restrict__ W2,
    unsigned short* __restrict__ xh, unsigned short* __restrict__ w1t,
    unsigned short* __restrict__ w2t) {
  int i = blockIdx.x * 256 + threadIdx.x;
  const int NX  = BATCH * DIMC;   // 262144
  const int NW1 = DIMC * EMB;     // 32768
  const int NW2 = EMB * EMB;      // 16384
  if (i < NX) { xh[i] = f2bf_bits(x[i]); return; }
  i -= NX;
  if (i < NW1) {                  // w1t[n*DIMC + k] = W1[k*EMB + n]
    int n = i / DIMC, k = i % DIMC;
    w1t[i] = f2bf_bits(W1[(size_t)k * EMB + n]);
    return;
  }
  i -= NW1;
  if (i < NW2) {                  // w2t[n*EMB + k] = W2[k*EMB + n]
    int n = i / EMB, k = i % EMB;
    w2t[i] = f2bf_bits(W2[(size_t)k * EMB + n]);
  }
}

// ---------- kernel 1: context MLP (relu(x@W1+b1)@W2 + b2), all-bf16 data path ----------

__global__ __launch_bounds__(32) void mlp_kernel(
    const unsigned short* __restrict__ xh, const float* __restrict__ b1,
    const unsigned short* __restrict__ w1t, const float* __restrict__ b2,
    const unsigned short* __restrict__ w2t, unsigned short* __restrict__ ctxh) {
  __shared__ unsigned short h[16 * EMB];   // bf16 hidden tile
  const int lane = threadIdx.x & 31;
  const int l  = lane & 15;
  const int hi = (lane >> 4) & 1;
  const int rowBase = blockIdx.x * 16;

  bf16x16 afr[8];
#pragma unroll
  for (int ks = 0; ks < 8; ++ks)
    afr[ks] = load_afrag_h(xh + (size_t)rowBase * DIMC, DIMC, ks * 32, lane);

  for (int nt = 0; nt < 8; ++nt) {
    floatx8 acc;
#pragma unroll
    for (int j = 0; j < 8; ++j) acc[j] = 0.0f;
#pragma unroll
    for (int ks = 0; ks < 8; ++ks)
      acc = wmma_bf16(afr[ks], load_bfrag_T(w1t, DIMC, nt * 16, ks * 32, lane, EMB), acc);
    const float bias = b1[nt * 16 + l];
#pragma unroll
    for (int r = 0; r < 8; ++r) {
      const int m = r + hi * 8;
      h[m * EMB + nt * 16 + l] = f2bf_bits(fmaxf(acc[r] + bias, 0.0f));
    }
  }
  __syncthreads();

  bf16x16 hfr[4];
#pragma unroll
  for (int ks = 0; ks < 4; ++ks) hfr[ks] = load_afrag_h(h, EMB, ks * 32, lane);

  for (int nt = 0; nt < 8; ++nt) {
    floatx8 acc;
#pragma unroll
    for (int j = 0; j < 8; ++j) acc[j] = 0.0f;
#pragma unroll
    for (int ks = 0; ks < 4; ++ks)
      acc = wmma_bf16(hfr[ks], load_bfrag_T(w2t, EMB, nt * 16, ks * 32, lane, EMB), acc);
    const float bias = b2[nt * 16 + l];
#pragma unroll
    for (int r = 0; r < 8; ++r) {
      const int m = r + hi * 8;
      ctxh[(size_t)(rowBase + m) * EMB + nt * 16 + l] = f2bf_bits(acc[r] + bias);
    }
  }
}

// ---------- kernel 2: WMMA scoring (32 rows/block) + chunk stats + chunk top-100 ----------

__global__ __launch_bounds__(128) void score_topk_kernel(
    const unsigned short* __restrict__ ctxh, const unsigned short* __restrict__ embh,
    float* __restrict__ chunkMax, float* __restrict__ chunkSum,
    float* __restrict__ candVal, int* __restrict__ candIdx) {
  __shared__ unsigned short slog[ROWS_PER_BLOCK * CHUNK];   // 128 KB bf16 logit tile
  __shared__ int   shist[4 * 256];
  __shared__ int   scnt[4];
#if HAS_ASYNC
  __shared__ unsigned short stage[4][2][16 * EMB];          // 32 KB double-buffered item tiles
#endif

  const int chunk   = blockIdx.x;
  const int rowBase = blockIdx.y * ROWS_PER_BLOCK;
  const int wv   = threadIdx.x >> 5;
  const int lane = threadIdx.x & 31;
  const int l    = lane & 15;
  const int hi   = (lane >> 4) & 1;
  const int chunkBase = chunk * CHUNK;

  // Hoist A (ctx) fragments: 2 M-tiles x 4 k-steps.
  bf16x16 afr[MTILES][4];
#pragma unroll
  for (int mt = 0; mt < MTILES; ++mt)
#pragma unroll
    for (int ks = 0; ks < 4; ++ks)
      afr[mt][ks] = load_afrag_h(ctxh + (size_t)(rowBase + mt * 16) * EMB, EMB, ks * 32, lane);

  const int itemsPerWave = CHUNK / 4;          // 512
  const int T = itemsPerWave / 16;             // 32 tiles per wave

#if HAS_ASYNC
  // Pipeline: async-stage tile t+1 into LDS while computing tile t.
  stage_tile_async(embh, chunkBase + wv * itemsPerWave, &stage[wv][0][0], lane);
  for (int t = 0; t < T; ++t) {
    const int localItem = wv * itemsPerWave + t * 16;
    const int cur = t & 1;
    if (t + 1 < T) {
      stage_tile_async(embh, chunkBase + localItem + 16, &stage[wv][cur ^ 1][0], lane);
      __builtin_amdgcn_s_wait_asynccnt(8);     // current tile's 8 copies complete (in-order)
    } else {
      __builtin_amdgcn_s_wait_asynccnt(0);
    }
    floatx8 acc[MTILES];
#pragma unroll
    for (int mt = 0; mt < MTILES; ++mt)
#pragma unroll
      for (int j = 0; j < 8; ++j) acc[mt][j] = 0.0f;
#pragma unroll
    for (int ks = 0; ks < 4; ++ks) {
      bf16x16 bb = load_bfrag_lds(&stage[wv][cur][0], ks * 32, lane);
#pragma unroll
      for (int mt = 0; mt < MTILES; ++mt)
        acc[mt] = wmma_bf16(afr[mt][ks], bb, acc[mt]);
    }
#pragma unroll
    for (int mt = 0; mt < MTILES; ++mt)
#pragma unroll
      for (int r = 0; r < 8; ++r) {
        const int m = mt * 16 + r + hi * 8;
        slog[m * CHUNK + localItem + l] = f2bf_bits(acc[mt][r]);
      }
  }
#else
  for (int t = 0; t < T; ++t) {
    const int localItem = wv * itemsPerWave + t * 16;
    const int nb = chunkBase + localItem;
    {
      int pref = nb + 16 + l;
      if (pref >= NITEMS) pref = NITEMS - 1;
      __builtin_prefetch(embh + (size_t)pref * EMB, 0, 1);
    }
    floatx8 acc[MTILES];
#pragma unroll
    for (int mt = 0; mt < MTILES; ++mt)
#pragma unroll
      for (int j = 0; j < 8; ++j) acc[mt][j] = 0.0f;
#pragma unroll
    for (int ks = 0; ks < 4; ++ks) {
      bf16x16 bb = load_bfrag_T(embh, EMB, nb, ks * 32, lane, NITEMS);
#pragma unroll
      for (int mt = 0; mt < MTILES; ++mt)
        acc[mt] = wmma_bf16(afr[mt][ks], bb, acc[mt]);
    }
#pragma unroll
    for (int mt = 0; mt < MTILES; ++mt)
#pragma unroll
      for (int r = 0; r < 8; ++r) {
        const int m = mt * 16 + r + hi * 8;
        slog[m * CHUNK + localItem + l] = f2bf_bits(acc[mt][r]);
      }
  }
#endif
  __syncthreads();

  // Mask out-of-range items (only last chunk) with bf16 -inf (0xFF80).
  int valid = NITEMS - chunkBase;
  if (valid > CHUNK) valid = CHUNK;
  for (int i = valid + (int)threadIdx.x; i < CHUNK; i += 128) {
#pragma unroll
    for (int m = 0; m < ROWS_PER_BLOCK; ++m) slog[m * CHUNK + i] = 0xFF80u;
  }
  __syncthreads();

  // Each wave handles 8 rows sequentially; barrier counts identical across waves.
  int* hist = &shist[wv * 256];
  for (int rr = 0; rr < 8; ++rr) {
    const int m    = wv * 8 + rr;
    const int grow = rowBase + m;
    const unsigned short* rl = &slog[m * CHUNK];

    float lm = -__builtin_inff();
    for (int i = lane; i < CHUNK; i += 32) lm = fmaxf(lm, bf2f(rl[i]));
#pragma unroll
    for (int o = 16; o > 0; o >>= 1) lm = fmaxf(lm, __shfl_xor(lm, o, 32));
    float ls = 0.0f;
    for (int i = lane; i < CHUNK; i += 32) ls += __expf(bf2f(rl[i]) - lm);
#pragma unroll
    for (int o = 16; o > 0; o >>= 1) ls += __shfl_xor(ls, o, 32);
    if (lane == 0) {
      chunkMax[(size_t)grow * NCHUNK + chunk] = lm;
      chunkSum[(size_t)grow * NCHUNK + chunk] = ls;
    }

    for (int b = lane; b < 256; b += 32) hist[b] = 0;
    __syncthreads();
    for (int i = lane; i < CHUNK; i += 32) atomicAdd(&hist[key16(rl[i]) >> 8], 1);
    __syncthreads();
    int b1 = 0, chi = 0;
    { int cum = 0; bool done = false;
      for (int b = 255; b >= 0; --b) { int c = hist[b];
        if (!done && cum + c >= TOPK) { b1 = b; chi = cum; done = true; } cum += c; } }
    __syncthreads();
    for (int b = lane; b < 256; b += 32) hist[b] = 0;
    __syncthreads();
    for (int i = lane; i < CHUNK; i += 32) {
      unsigned k = key16(rl[i]);
      if ((int)(k >> 8) == b1) atomicAdd(&hist[k & 255], 1);
    }
    __syncthreads();
    int b2 = 0;
    { int cum2 = 0; bool done = false;
      for (int b = 255; b >= 0; --b) { int c = hist[b];
        if (!done && chi + cum2 + c >= TOPK) { b2 = b; done = true; } cum2 += c; } }
    const unsigned thr = ((unsigned)b1 << 8) | (unsigned)b2;

    if (lane == 0) scnt[wv] = 0;
    __syncthreads();
    const size_t base = ((size_t)grow * NCHUNK + chunk) * TOPK;
    for (int i = lane; i < CHUNK; i += 32) {
      unsigned short s = rl[i];
      if (key16(s) > thr) {
        int p = atomicAdd(&scnt[wv], 1);
        if (p < TOPK) { candVal[base + p] = bf2f(s); candIdx[base + p] = chunkBase + i; }
      }
    }
    __syncthreads();
    for (int i = lane; i < CHUNK; i += 32) {
      unsigned short s = rl[i];
      if (key16(s) == thr) {
        int p = atomicAdd(&scnt[wv], 1);
        if (p < TOPK) { candVal[base + p] = bf2f(s); candIdx[base + p] = chunkBase + i; }
      }
    }
    __syncthreads();
  }
}

// ---------- kernel 3: merge chunk stats + global top-100 + softmax + sort ----------

__global__ __launch_bounds__(256) void merge_kernel(
    const float* __restrict__ chunkMax, const float* __restrict__ chunkSum,
    const float* __restrict__ candVal, const int* __restrict__ candIdx,
    float* __restrict__ outProb, float* __restrict__ outIdx) {
  __shared__ int   hist[256];
  __shared__ float red[256];
  __shared__ float fval[TOPK];
  __shared__ int   fidx[TOPK];
  __shared__ int   cnt;

  const int row = blockIdx.x;
  const int tid = threadIdx.x;

  float lm = -__builtin_inff();
  for (int c = tid; c < NCHUNK; c += 256) lm = fmaxf(lm, chunkMax[(size_t)row * NCHUNK + c]);
  red[tid] = lm;
  __syncthreads();
  for (int s = 128; s > 0; s >>= 1) { if (tid < s) red[tid] = fmaxf(red[tid], red[tid + s]); __syncthreads(); }
  const float gmax = red[0];
  __syncthreads();

  float lsum = 0.0f;
  for (int c = tid; c < NCHUNK; c += 256)
    lsum += chunkSum[(size_t)row * NCHUNK + c] * __expf(chunkMax[(size_t)row * NCHUNK + c] - gmax);
  red[tid] = lsum;
  __syncthreads();
  for (int s = 128; s > 0; s >>= 1) { if (tid < s) red[tid] += red[tid + s]; __syncthreads(); }
  const float denom = red[0];
  __syncthreads();

  const int M = NCHUNK * TOPK;
  const float* cv = candVal + (size_t)row * M;
  const int*   ci = candIdx + (size_t)row * M;

  hist[tid] = 0;
  __syncthreads();
  for (int i = tid; i < M; i += 256) atomicAdd(&hist[fkey(cv[i]) >> 24], 1);
  __syncthreads();
  int b1 = 0, chi = 0;
  { int cum = 0; bool done = false;
    for (int b = 255; b >= 0; --b) { int c = hist[b];
      if (!done && cum + c >= TOPK) { b1 = b; chi = cum; done = true; } cum += c; } }
  __syncthreads();
  hist[tid] = 0;
  __syncthreads();
  for (int i = tid; i < M; i += 256) { unsigned k = fkey(cv[i]);
    if ((int)(k >> 24) == b1) atomicAdd(&hist[(k >> 16) & 255], 1); }
  __syncthreads();
  int b2 = 0;
  { int cum2 = 0; bool done = false;
    for (int b = 255; b >= 0; --b) { int c = hist[b];
      if (!done && chi + cum2 + c >= TOPK) { b2 = b; done = true; } cum2 += c; } }
  const unsigned thr = ((unsigned)b1 << 8) | (unsigned)b2;

  if (tid == 0) cnt = 0;
  __syncthreads();
  for (int i = tid; i < M; i += 256) { float v = cv[i];
    if ((fkey(v) >> 16) > thr) { int p = atomicAdd(&cnt, 1); if (p < TOPK) { fval[p] = v; fidx[p] = ci[i]; } } }
  __syncthreads();
  for (int i = tid; i < M; i += 256) { float v = cv[i];
    if ((fkey(v) >> 16) == thr) { int p = atomicAdd(&cnt, 1); if (p < TOPK) { fval[p] = v; fidx[p] = ci[i]; } } }
  __syncthreads();

  if (tid < TOPK) {
    const float v  = fval[tid];
    const int   id = fidx[tid];
    int rank = 0;
    for (int j = 0; j < TOPK; ++j) {
      float vj = fval[j];
      rank += (vj > v) || (vj == v && fidx[j] < id);
    }
    outProb[(size_t)row * TOPK + rank] = __expf(v - gmax) / denom;
    outIdx[(size_t)row * TOPK + rank]  = (float)id;
  }
}

// ---------- launcher ----------

extern "C" void kernel_launch(void* const* d_in, const int* in_sizes, int n_in,
                              void* d_out, int out_size, void* d_ws, size_t ws_size,
                              hipStream_t stream) {
  (void)in_sizes; (void)n_in; (void)out_size; (void)ws_size;
  const float* x   = (const float*)d_in[0];
  const float* W1  = (const float*)d_in[1];
  const float* b1  = (const float*)d_in[2];
  const float* W2  = (const float*)d_in[3];
  const float* b2  = (const float*)d_in[4];
  const float* emb = (const float*)d_in[5];

  char* ws = (char*)d_ws;
  unsigned short* embh = (unsigned short*)ws;                       // 64,000,000 B
  ws += (size_t)NITEMS * EMB * sizeof(unsigned short);
  unsigned short* ctxh = (unsigned short*)ws;                       // 262,144 B
  ws += (size_t)BATCH * EMB * sizeof(unsigned short);
  unsigned short* xh = (unsigned short*)ws;                         // 524,288 B
  ws += (size_t)BATCH * DIMC * sizeof(unsigned short);
  unsigned short* w1t = (unsigned short*)ws;                        // 65,536 B
  ws += (size_t)DIMC * EMB * sizeof(unsigned short);
  unsigned short* w2t = (unsigned short*)ws;                        // 32,768 B
  ws += (size_t)EMB * EMB * sizeof(unsigned short);
  float* cVal = (float*)ws;                                         // 50,380,800 B
  ws += (size_t)BATCH * NCHUNK * TOPK * sizeof(float);
  int* cIdx = (int*)ws;                                             // 50,380,800 B
  ws += (size_t)BATCH * NCHUNK * TOPK * sizeof(int);
  float* cMax = (float*)ws;
  ws += (size_t)BATCH * NCHUNK * sizeof(float);
  float* cSum = (float*)ws;

  float* outP = (float*)d_out;                                      // [1024,100] probs
  float* outI = outP + (size_t)BATCH * TOPK;                        // [1024,100] indices

  cvt_emb_kernel<<<(NITEMS * EMB) / (256 * 8), 256, 0, stream>>>(emb, embh);
  cvt_params_kernel<<<(BATCH * DIMC + DIMC * EMB + EMB * EMB + 255) / 256, 256, 0, stream>>>(
      x, W1, W2, xh, w1t, w2t);
  mlp_kernel<<<BATCH / 16, 32, 0, stream>>>(xh, b1, w1t, b2, w2t, ctxh);
  score_topk_kernel<<<dim3(NCHUNK, ROWBLOCKS), 128, 0, stream>>>(ctxh, embh, cMax, cSum, cVal, cIdx);
  merge_kernel<<<BATCH, 256, 0, stream>>>(cMax, cSum, cVal, cIdx, outP, outI);
}
